// Language_73349451481501
// MI455X (gfx1250) — compile-verified, matrix operand order checked by reading, not run on hardware
//
#include <hip/hip_runtime.h>
#include <hip/hip_bf16.h>
#include <stdint.h>

// ---------------------------------------------------------------------------
// Problem constants
// ---------------------------------------------------------------------------
#define BATCH 2048
#define TSTEP 30
#define HDIM  512
#define DDIM  512
#define VDIM  6207
#define XDIM  2048
#define GDIM  2048          // 4*H
#define KCAT  1024          // D + H (== H + H for top cell)

// GEMM tiling
#define BM 128
#define BN 128
#define BK 32
#define NKT (KCAT / BK)     // 32 k-iterations
#define KSPLIT (DDIM / BK)  // 16: uniform boundary between the two A regions
#define LDT 40              // padded LDS row stride in halves (80B, 16B aligned)

typedef __attribute__((ext_vector_type(16))) _Float16 v16h;
typedef __attribute__((ext_vector_type(8)))  float    v8f;

struct Chunks32 { uint4 lo, hi; };

__device__ __forceinline__ float sigf(float x) { return 1.0f / (1.0f + __expf(-x)); }

// deterministic hash -> [0,1)
__device__ __forceinline__ float uhash(uint32_t a, uint32_t b) {
    uint32_t s = a * 0x9E3779B9u ^ (b + 0x7F4A7C15u) * 0x85EBCA6Bu;
    s ^= s >> 16; s *= 0x7FEB352Du;
    s ^= s >> 15; s *= 0x846CA68Bu;
    s ^= s >> 16;
    return (float)(s >> 8) * (1.0f / 16777216.0f);
}

// Per-lane WMMA fragment load from an LDS tile laid out [row][LDT] halves.
// ISA wave32 16-bit 16x32 A/B layout: lane&15 selects row/col, kb=(lane>>4)*8
// selects K sub-chunks {kb..kb+7} and {kb+16..kb+23}.
__device__ __forceinline__ v16h frag_ld(const _Float16* tile, int row, int kb) {
    const _Float16* p = tile + row * LDT;
    Chunks32 c;
    c.lo = *(const uint4*)(p + kb);
    c.hi = *(const uint4*)(p + kb + 16);
    return __builtin_bit_cast(v16h, c);
}

__device__ __forceinline__ uint4 cvt8_f32_to_f16(const float* __restrict__ p) {
    float4 f0 = *(const float4*)p;
    float4 f1 = *(const float4*)(p + 4);
    union { _Float16 h[8]; uint4 u; } pk;
    pk.h[0] = (_Float16)f0.x; pk.h[1] = (_Float16)f0.y;
    pk.h[2] = (_Float16)f0.z; pk.h[3] = (_Float16)f0.w;
    pk.h[4] = (_Float16)f1.x; pk.h[5] = (_Float16)f1.y;
    pk.h[6] = (_Float16)f1.z; pk.h[7] = (_Float16)f1.w;
    return pk.u;
}

__device__ __forceinline__ uint4 sel4(bool keep, uint4 v) {
    uint4 r;
    r.x = keep ? v.x : 0u; r.y = keep ? v.y : 0u;
    r.z = keep ? v.z : 0u; r.w = keep ? v.w : 0u;
    return r;
}

// ---------------------------------------------------------------------------
// Fused gates GEMM:  G[B][4H] = Acat[B][1024] x Wcat16^T   (Wcat16: [4H][1024] f16)
// MODE 0: Acat = [ emb16[y[b,t]] (k<512) ; h_f32 (k>=512, cvt) ]
// MODE 1: Acat = [ gate[b] ? h16 : 0     ; ht16            ]
// Double-buffered LDS, one barrier per K-iteration.
// ---------------------------------------------------------------------------
template <int MODE>
__global__ __launch_bounds__(256)
void gemm_gates(const _Float16* __restrict__ W16,
                const _Float16* __restrict__ src16,   // emb16 (MODE0) / h16 (MODE1)
                const float*    __restrict__ hsrc,    // h f32 (MODE0)
                const _Float16* __restrict__ ht16,    // (MODE1)
                const float*    __restrict__ gatev,   // (MODE1)
                const int*      __restrict__ y,       // (MODE0)
                int t,
                float* __restrict__ G) {
    __shared__ _Float16 sA[2][BM * LDT];
    __shared__ _Float16 sB[2][BN * LDT];

    const int tid  = threadIdx.x;
    const int bm   = blockIdx.y * BM;
    const int bn   = blockIdx.x * BN;
    const int wave = tid >> 5;
    const int lane = tid & 31;
    const int wm   = (wave >> 1) * 32;     // 4 wave-rows of 32
    const int wn   = (wave & 1) * 64;      // 2 wave-cols of 64
    const int rr   = lane & 15;
    const int kb   = (lane >> 4) * 8;

    // ---- loop-invariant loader state: each thread owns 2 A-chunks + 2 B-chunks
    const int lr  = tid >> 2;              // 0..63
    const int lkc = (tid & 3) * 8;         // 0,8,16,24 (halves within BK)
    const int rA0 = bm + lr, rA1 = rA0 + 64;
    const int rB0 = bn + lr, rB1 = rB0 + 64;

    // region-0 source rows (emb gather or h16), region-1 source rows (h f32 or ht16)
    const _Float16* a0r0;
    const _Float16* a1r0;
    bool keep0 = true, keep1 = true;
    if (MODE == 0) {
        a0r0 = src16 + (size_t)y[rA0 * TSTEP + t] * DDIM + lkc;
        a1r0 = src16 + (size_t)y[rA1 * TSTEP + t] * DDIM + lkc;
    } else {
        a0r0 = src16 + (size_t)rA0 * HDIM + lkc;
        a1r0 = src16 + (size_t)rA1 * HDIM + lkc;
        keep0 = gatev[rA0] != 0.0f;
        keep1 = gatev[rA1] != 0.0f;
    }
    const float*    hA0 = hsrc + (size_t)rA0 * HDIM + lkc;   // MODE0 region 1
    const float*    hA1 = hsrc + (size_t)rA1 * HDIM + lkc;
    const _Float16* tA0 = ht16 + (size_t)rA0 * HDIM + lkc;   // MODE1 region 1
    const _Float16* tA1 = ht16 + (size_t)rA1 * HDIM + lkc;
    const _Float16* wB0 = W16 + (size_t)rB0 * KCAT + lkc;
    const _Float16* wB1 = W16 + (size_t)rB1 * KCAT + lkc;

    _Float16* stA0 = &sA[0][lr * LDT + lkc];
    _Float16* stA1 = &sA[0][(lr + 64) * LDT + lkc];
    _Float16* stB0 = &sB[0][lr * LDT + lkc];
    _Float16* stB1 = &sB[0][(lr + 64) * LDT + lkc];
    const size_t bufStride = (size_t)BM * LDT;   // halves per buffer

    v8f acc[2][4] = {};

    // fetch tile kt into registers (kt is uniform; region branch is scalar)
    uint4 ra0, ra1, rb0, rb1;
    auto fetch = [&](int kt) {
        const int k0 = kt * BK;             // region-0 offset (halves)
        const int k1 = (kt - KSPLIT) * BK;  // region-1 offset
        if (kt < KSPLIT) {
            if (MODE == 0) {
                ra0 = *(const uint4*)(a0r0 + k0);
                ra1 = *(const uint4*)(a1r0 + k0);
            } else {
                ra0 = sel4(keep0, *(const uint4*)(a0r0 + k0));
                ra1 = sel4(keep1, *(const uint4*)(a1r0 + k0));
            }
        } else {
            if (MODE == 0) {
                ra0 = cvt8_f32_to_f16(hA0 + k1);
                ra1 = cvt8_f32_to_f16(hA1 + k1);
            } else {
                ra0 = *(const uint4*)(tA0 + k1);
                ra1 = *(const uint4*)(tA1 + k1);
            }
        }
        rb0 = *(const uint4*)(wB0 + k0);
        rb1 = *(const uint4*)(wB1 + k0);
    };
    auto stage = [&](int buf) {
        const size_t o = (size_t)buf * bufStride;
        *(uint4*)(stA0 + o) = ra0;
        *(uint4*)(stA1 + o) = ra1;
        *(uint4*)(stB0 + o) = rb0;
        *(uint4*)(stB1 + o) = rb1;
    };

    fetch(0);
    stage(0);

    for (int kt = 0; kt < NKT; ++kt) {
        __syncthreads();                    // buf[kt&1] ready; buf[1-(kt&1)] free
        const int cur = kt & 1;
        const bool more = (kt + 1) < NKT;
        if (more) fetch(kt + 1);
        if (kt + 2 < NKT)                   // prefetch weight stream 2 tiles ahead
            __builtin_prefetch(wB0 + (kt + 2) * BK, 0, 1);

        const _Float16* tA = &sA[cur][0];
        const _Float16* tB = &sB[cur][0];
        v16h a0 = frag_ld(tA, wm + rr,      kb);
        v16h a1 = frag_ld(tA, wm + 16 + rr, kb);
        v16h b0 = frag_ld(tB, wn + rr,      kb);
        v16h b1 = frag_ld(tB, wn + 16 + rr, kb);
        v16h b2 = frag_ld(tB, wn + 32 + rr, kb);
        v16h b3 = frag_ld(tB, wn + 48 + rr, kb);

        acc[0][0] = __builtin_amdgcn_wmma_f32_16x16x32_f16(false, a0, false, b0, (short)0, acc[0][0], false, false);
        acc[0][1] = __builtin_amdgcn_wmma_f32_16x16x32_f16(false, a0, false, b1, (short)0, acc[0][1], false, false);
        acc[0][2] = __builtin_amdgcn_wmma_f32_16x16x32_f16(false, a0, false, b2, (short)0, acc[0][2], false, false);
        acc[0][3] = __builtin_amdgcn_wmma_f32_16x16x32_f16(false, a0, false, b3, (short)0, acc[0][3], false, false);
        acc[1][0] = __builtin_amdgcn_wmma_f32_16x16x32_f16(false, a1, false, b0, (short)0, acc[1][0], false, false);
        acc[1][1] = __builtin_amdgcn_wmma_f32_16x16x32_f16(false, a1, false, b1, (short)0, acc[1][1], false, false);
        acc[1][2] = __builtin_amdgcn_wmma_f32_16x16x32_f16(false, a1, false, b2, (short)0, acc[1][2], false, false);
        acc[1][3] = __builtin_amdgcn_wmma_f32_16x16x32_f16(false, a1, false, b3, (short)0, acc[1][3], false, false);

        if (more) stage(1 - cur);
    }

    // epilogue: D layout — lanes<16: M=r, lanes>=16: M=8+r ; N=lane&15
    const int mbase = bm + wm + (lane >> 4) * 8;
    const int nbase = bn + wn + (lane & 15);
    for (int mi = 0; mi < 2; ++mi)
        for (int ni = 0; ni < 4; ++ni)
            for (int r = 0; r < 8; ++r)
                G[(size_t)(mbase + mi * 16 + r) * GDIM + nbase + ni * 16] = acc[mi][ni][r];
}

// ---------------------------------------------------------------------------
// Elementwise LSTM cells + gate + prep kernels
// ---------------------------------------------------------------------------
__global__ void cell_bottom(const float* __restrict__ G,
                            const float* __restrict__ bih, const float* __restrict__ bhh,
                            float* __restrict__ h, float* __restrict__ c,
                            _Float16* __restrict__ h16) {
    int idx = blockIdx.x * blockDim.x + threadIdx.x;   // B*H
    int b = idx >> 9, j = idx & 511;
    const float* g = G + (size_t)b * GDIM;
    float gi = g[j]          + bih[j]          + bhh[j];
    float gf = g[j + 512]    + bih[j + 512]    + bhh[j + 512];
    float gg = g[j + 1024]   + bih[j + 1024]   + bhh[j + 1024];
    float go = g[j + 1536]   + bih[j + 1536]   + bhh[j + 1536];
    float cn = sigf(gf) * c[idx] + sigf(gi) * tanhf(gg);
    float hn = sigf(go) * tanhf(cn);
    c[idx] = cn; h[idx] = hn; h16[idx] = (_Float16)hn;
}

__global__ __launch_bounds__(256)
void gate_kernel(const float* __restrict__ h, const float* __restrict__ wvec,
                 const float* __restrict__ xg, const float* __restrict__ consts,
                 int t, float* __restrict__ gate) {
    int lane = threadIdx.x & 31;
    int b = blockIdx.x * 8 + (threadIdx.x >> 5);
    const float* hr = h + (size_t)b * HDIM;
    float acc = 0.0f;
    for (int j = lane; j < HDIM; j += 32) acc += hr[j] * wvec[j];
    for (int o = 16; o; o >>= 1) acc += __shfl_xor(acc, o, 32);
    if (lane == 0) {
        float logit = acc + xg[b] + consts[0];
        float U  = uhash((uint32_t)t * 2654435761u + 0x1234567u, (uint32_t)b);
        float nz = __logf(U + 1e-10f) - __logf(1.0f - U + 1e-10f);
        float ys = sigf((logit + nz) * (1.0f / 0.3f));
        gate[b] = (ys >= 0.5f) ? 1.0f : 0.0f;
    }
}

__global__ void cell_top(const float* __restrict__ G,
                         const float* __restrict__ bih, const float* __restrict__ bhh,
                         const float* __restrict__ gatev,
                         float* __restrict__ ht, float* __restrict__ ct,
                         _Float16* __restrict__ ht16, float* __restrict__ htsum) {
    int idx = blockIdx.x * blockDim.x + threadIdx.x;
    int b = idx >> 9, j = idx & 511;
    const float* g = G + (size_t)b * GDIM;
    float gi = g[j]          + bih[j]          + bhh[j];
    float gf = g[j + 512]    + bih[j + 512]    + bhh[j + 512];
    float gg = g[j + 1024]   + bih[j + 1024]   + bhh[j + 1024];
    float go = g[j + 1536]   + bih[j + 1536]   + bhh[j + 1536];
    float ctb = sigf(gf) * ct[idx] + sigf(gi) * tanhf(gg);
    float htb = sigf(go) * tanhf(ctb);
    float gv = gatev[b];
    float ctn = gv * ctb + (1.0f - gv) * ct[idx];
    float htn = gv * htb + (1.0f - gv) * ht[idx];
    ct[idx] = ctn; ht[idx] = htn;
    ht16[idx] = (_Float16)htn;
    htsum[idx] += htn;
}

__global__ void conv_weights(const float* __restrict__ Wih1, const float* __restrict__ Whh1,
                             const float* __restrict__ Wih2, const float* __restrict__ Whh2,
                             _Float16* __restrict__ W1, _Float16* __restrict__ W2) {
    int idx = blockIdx.x * blockDim.x + threadIdx.x;   // 4H * KCAT
    int n = idx >> 10, k = idx & 1023;
    float a = (k < DDIM) ? Wih1[(size_t)n * DDIM + k] : Whh1[(size_t)n * HDIM + (k - DDIM)];
    float b = (k < HDIM) ? Wih2[(size_t)n * HDIM + k] : Whh2[(size_t)n * HDIM + (k - HDIM)];
    W1[idx] = (_Float16)a; W2[idx] = (_Float16)b;
}

__global__ void conv_emb(const float* __restrict__ emb, _Float16* __restrict__ emb16) {
    int idx = blockIdx.x * blockDim.x + threadIdx.x;   // V*D
    emb16[idx] = (_Float16)emb[idx];
}

// u[j]    = sum_i Wx[i][j] * Wg[512+i]   (XD outputs)
// wvec[j] = sum_i Wg[i]    * Wh[i][j]    (H outputs)
__global__ void vec_kernel(const float* __restrict__ Wx, const float* __restrict__ Wg,
                           const float* __restrict__ Wh,
                           float* __restrict__ u, float* __restrict__ wvec) {
    int idx = blockIdx.x * blockDim.x + threadIdx.x;   // 2560
    if (idx < XDIM) {
        float acc = 0.0f;
        for (int i = 0; i < XDIM; ++i) acc += Wx[(size_t)i * XDIM + idx] * Wg[HDIM + i];
        u[idx] = acc;
    } else {
        int j = idx - XDIM;
        float acc = 0.0f;
        for (int i = 0; i < HDIM; ++i) acc += Wg[i] * Wh[(size_t)i * HDIM + j];
        wvec[j] = acc;
    }
}

__global__ void consts_kernel(const float* __restrict__ Wg, const float* __restrict__ bh,
                              const float* __restrict__ bx, const float* __restrict__ bg,
                              float* __restrict__ consts) {
    __shared__ float red[256];
    int tid = threadIdx.x;
    float a = 0.0f, b = 0.0f;
    for (int i = tid; i < HDIM; i += 256) a += Wg[i] * bh[i];
    for (int i = tid; i < XDIM; i += 256) b += Wg[HDIM + i] * bx[i];
    red[tid] = a; __syncthreads();
    for (int s = 128; s; s >>= 1) { if (tid < s) red[tid] += red[tid + s]; __syncthreads(); }
    float ch = red[0]; __syncthreads();
    red[tid] = b; __syncthreads();
    for (int s = 128; s; s >>= 1) { if (tid < s) red[tid] += red[tid + s]; __syncthreads(); }
    if (tid == 0) { consts[0] = ch + bg[0]; consts[1] = red[0]; }
}

__global__ __launch_bounds__(256)
void xg_kernel(const float* __restrict__ x, const float* __restrict__ u,
               const float* __restrict__ consts, float* __restrict__ xg) {
    int lane = threadIdx.x & 31;
    int b = blockIdx.x * 8 + (threadIdx.x >> 5);
    const float* xr = x + (size_t)b * XDIM;
    float acc = 0.0f;
    for (int j = lane; j < XDIM; j += 32) acc += xr[j] * u[j];
    for (int o = 16; o; o >>= 1) acc += __shfl_xor(acc, o, 32);
    if (lane == 0) xg[b] = acc + consts[1];
}

__global__ void init_state(float* __restrict__ h, float* __restrict__ c,
                           float* __restrict__ ht, float* __restrict__ ct,
                           _Float16* __restrict__ h16, _Float16* __restrict__ ht16,
                           float* __restrict__ htsum) {
    int idx = blockIdx.x * blockDim.x + threadIdx.x;   // B*H
    c[idx]  = uhash(0xC0FFEEu,  (uint32_t)idx) * 0.2f - 0.1f;
    ct[idx] = uhash(0xBADF00Du, (uint32_t)idx) * 0.2f - 0.1f;
    h[idx] = 0.0f; ht[idx] = 0.0f;
    h16[idx] = (_Float16)0.0f; ht16[idx] = (_Float16)0.0f;
    htsum[idx] = 0.0f;
}

__global__ void final_mean(const float* __restrict__ htsum, float* __restrict__ out) {
    int idx = blockIdx.x * blockDim.x + threadIdx.x;
    out[idx] = htsum[idx] * (1.0f / (float)TSTEP);
}

// ---------------------------------------------------------------------------
// Host launcher
// ---------------------------------------------------------------------------
extern "C" void kernel_launch(void* const* d_in, const int* in_sizes, int n_in,
                              void* d_out, int out_size, void* d_ws, size_t ws_size,
                              hipStream_t stream) {
    (void)in_sizes; (void)n_in; (void)out_size; (void)ws_size;

    const float* x        = (const float*)d_in[0];
    const int*   y        = (const int*)  d_in[1];
    const float* emb      = (const float*)d_in[2];
    const float* lstm_Wih = (const float*)d_in[3];
    const float* lstm_Whh = (const float*)d_in[4];
    const float* lstm_bih = (const float*)d_in[5];
    const float* lstm_bhh = (const float*)d_in[6];
    const float* top_Wih  = (const float*)d_in[7];
    const float* top_Whh  = (const float*)d_in[8];
    const float* top_bih  = (const float*)d_in[9];
    const float* top_bhh  = (const float*)d_in[10];
    const float* Wg       = (const float*)d_in[11];
    const float* bg       = (const float*)d_in[12];
    const float* Wh       = (const float*)d_in[13];
    const float* bh       = (const float*)d_in[14];
    const float* Wx       = (const float*)d_in[15];
    const float* bx       = (const float*)d_in[16];
    float* out = (float*)d_out;

    // workspace carve-up (all 16B aligned)
    char* w = (char*)d_ws;
    size_t off = 0;
    auto take = [&](size_t bytes) { char* p = w + off; off += (bytes + 15) & ~size_t(15); return p; };
    _Float16* W1    = (_Float16*)take((size_t)GDIM * KCAT * 2);       // 4 MB
    _Float16* W2    = (_Float16*)take((size_t)GDIM * KCAT * 2);       // 4 MB
    _Float16* emb16 = (_Float16*)take((size_t)VDIM * DDIM * 2);       // ~6 MB
    float*    hbuf  = (float*)take((size_t)BATCH * HDIM * 4);
    float*    cbuf  = (float*)take((size_t)BATCH * HDIM * 4);
    float*    htbuf = (float*)take((size_t)BATCH * HDIM * 4);
    float*    ctbuf = (float*)take((size_t)BATCH * HDIM * 4);
    float*    htsum = (float*)take((size_t)BATCH * HDIM * 4);
    _Float16* h16   = (_Float16*)take((size_t)BATCH * HDIM * 2);
    _Float16* ht16  = (_Float16*)take((size_t)BATCH * HDIM * 2);
    float*    G     = (float*)take((size_t)BATCH * GDIM * 4);         // 16 MB
    float*    gate  = (float*)take((size_t)BATCH * 4);
    float*    xg    = (float*)take((size_t)BATCH * 4);
    float*    uvec  = (float*)take((size_t)XDIM * 4);
    float*    wvec  = (float*)take((size_t)HDIM * 4);
    float*    cns   = (float*)take(16);

    const int BH = BATCH * HDIM;

    // ---- prep (runs every call; deterministic) ----
    conv_weights<<<(GDIM * KCAT) / 256, 256, 0, stream>>>(lstm_Wih, lstm_Whh, top_Wih, top_Whh, W1, W2);
    conv_emb<<<(VDIM * DDIM) / 256, 256, 0, stream>>>(emb, emb16);
    vec_kernel<<<(XDIM + HDIM) / 256, 256, 0, stream>>>(Wx, Wg, Wh, uvec, wvec);
    consts_kernel<<<1, 256, 0, stream>>>(Wg, bh, bx, bg, cns);
    xg_kernel<<<BATCH / 8, 256, 0, stream>>>(x, uvec, cns, xg);
    init_state<<<BH / 256, 256, 0, stream>>>(hbuf, cbuf, htbuf, ctbuf, h16, ht16, htsum);

    dim3 ggrid(GDIM / BN, BATCH / BM);   // 16 x 16

    // ---- T recurrent steps ----
    for (int t = 0; t < TSTEP; ++t) {
        gemm_gates<0><<<ggrid, 256, 0, stream>>>(W1, emb16, hbuf, ht16, gate, y, t, G);
        cell_bottom<<<BH / 256, 256, 0, stream>>>(G, lstm_bih, lstm_bhh, hbuf, cbuf, h16);
        gate_kernel<<<BATCH / 8, 256, 0, stream>>>(hbuf, wvec, xg, cns, t, gate);
        gemm_gates<1><<<ggrid, 256, 0, stream>>>(W2, h16, hbuf, ht16, gate, y, t, G);
        cell_top<<<BH / 256, 256, 0, stream>>>(G, top_bih, top_bhh, gate, htbuf, ctbuf, ht16, htsum);
    }

    final_mean<<<BH / 256, 256, 0, stream>>>(htsum, out);
}
